// MN4_80444737454121
// MI455X (gfx1250) — matrix operand order repaired and checked by reading, not compiled, not run on hardware
//
#include <hip/hip_runtime.h>
#include <math.h>

// ---------------- problem constants (from reference) ----------------
#define BB    4      // batch
#define QQ    75     // queries per episode
#define CC    640    // channels (K dim)
#define HW    121    // 11*11 descriptors
#define NWAY  5
#define KSHOT 5
#define MS    605    // KSHOT*HW support descriptors per class
#define MSP   608    // per-class padded to 38 tiles of 16
#define MT    8      // 16-row M tiles per query (121 -> 128 rows)
#define KST   20     // 640 / 32 k-steps
#define JT    38     // MSP / 16 column tiles per class
#define NTILE (NWAY * JT)   // 190 column tiles total per (b,q)
#define FRAG  512    // halfs per WMMA fragment (32 lanes * 16 halfs)
#define TILE_HALFS (KST * FRAG)          // 10240 halfs = 20 KB per column tile
#define NEG_BIG (-1.0e30f)

typedef __attribute__((ext_vector_type(16))) _Float16 v16h;
typedef __attribute__((ext_vector_type(8)))  float    v8f;

// insert v into descending triple (t0 >= t1 >= t2)
__device__ __forceinline__ void top3_insert(float v, float& t0, float& t1, float& t2) {
    if (v > t0)      { t2 = t1; t1 = t0; t0 = v; }
    else if (v > t1) { t2 = t1; t1 = v; }
    else if (v > t2) { t2 = v; }
}

// ---------------- normalization scales ----------------
__global__ void knorm_q(const float* __restrict__ qf, float* __restrict__ qscale) {
    int bq = blockIdx.x;            // b*QQ + q
    int m  = threadIdx.x;           // 0..127
    float s = 0.f;
    if (m < HW) {
        const float* p = qf + (size_t)bq * CC * HW + m;
        for (int c = 0; c < CC; ++c) { float v = p[(size_t)c * HW]; s += v * v; }
    }
    qscale[bq * 128 + m] = (m < HW) ? (1.0f / (sqrtf(s) + 1e-8f)) : 0.0f;
}

__global__ void knorm_s(const float* __restrict__ sf, float* __restrict__ sscale) {
    int bs = blockIdx.x;                    // b*25 + n*5 + shot
    int b  = bs / (NWAY * KSHOT);
    int rem = bs % (NWAY * KSHOT);
    int n = rem / KSHOT, shot = rem % KSHOT;
    int p = threadIdx.x;
    if (p >= HW) return;
    const float* base = sf + (size_t)bs * CC * HW + p;
    float s = 0.f;
    for (int c = 0; c < CC; ++c) { float v = base[(size_t)c * HW]; s += v * v; }
    sscale[(b * NWAY + n) * MSP + shot * HW + p] = 1.0f / (sqrtf(s) + 1e-8f);
}

// ---------------- fragment packing ----------------
// A fragment (16-bit 16x32, ISA 7.12.2): lane L holds row M=L%16.
// lanes 0-15:  e0..7 -> K 0..7,   e8..15 -> K 16..23
// lanes 16-31: e0..7 -> K 8..15,  e8..15 -> K 24..31
__global__ void kpackA(const float* __restrict__ qf, const float* __restrict__ qscale,
                       _Float16* __restrict__ Apack) {
    int tid  = blockIdx.x * blockDim.x + threadIdx.x;
    int lane = tid & 31;
    int fa   = tid >> 5;                      // ((bq*MT + mtile)*KST + kstep)
    int kstep = fa % KST;
    int tmp   = fa / KST;
    int mtile = tmp % MT;
    int bq    = tmp / MT;
    int m = mtile * 16 + (lane & 15);
    v16h o;
    if (m < HW) {
        float sc = qscale[bq * 128 + m];
        const float* base = qf + (size_t)bq * CC * HW + m;
        int khalf = (lane & 16) ? 8 : 0;
#pragma unroll
        for (int e = 0; e < 16; ++e) {
            int k = kstep * 32 + khalf + ((e < 8) ? e : (8 + e));
            o[e] = (_Float16)(base[(size_t)k * HW] * sc);
        }
    } else {
#pragma unroll
        for (int e = 0; e < 16; ++e) o[e] = (_Float16)0.f;
    }
    *(v16h*)(Apack + (size_t)fa * FRAG + lane * 16) = o;
}

// B fragment (16-bit 32x16): lane L<16 holds col N=L, e0..7 -> K 0..7, e8..15 -> K 16..23;
// lane L>=16 holds col N=L-16, e0..7 -> K 8..15, e8..15 -> K 24..31.
__global__ void kpackB(const float* __restrict__ sf, const float* __restrict__ sscale,
                       _Float16* __restrict__ Bpack) {
    int tid  = blockIdx.x * blockDim.x + threadIdx.x;
    int lane = tid & 31;
    int fb   = tid >> 5;                      // ((bn*JT + jt)*KST + kstep)
    int kstep = fb % KST;
    int tmp   = fb / KST;
    int jt    = tmp % JT;
    int bn    = tmp / JT;
    int b = bn / NWAY, n = bn % NWAY;
    int col = jt * 16 + (lane & 15);
    v16h o;
    if (col < MS) {
        float sc = sscale[bn * MSP + col];
        int shot = col / HW, p = col % HW;
        const float* base = sf + (size_t)((b * NWAY + n) * KSHOT + shot) * CC * HW + p;
        int kbase = kstep * 32 + ((lane & 16) ? 8 : 0);
#pragma unroll
        for (int e = 0; e < 16; ++e)
            o[e] = (_Float16)(base[(size_t)(kbase + ((e < 8) ? e : (8 + e))) * HW] * sc);
    } else {
#pragma unroll
        for (int e = 0; e < 16; ++e) o[e] = (_Float16)0.f;
    }
    *(v16h*)(Bpack + (size_t)fb * FRAG + lane * 16) = o;
}

// ---------------- fused GEMM + reductions ----------------
// 1 block per (b,q); wave w computes M-tile w. B column tiles are shared by all
// 8 waves, staged once per block into LDS via async-to-LDS copies (ASYNCcnt),
// triple buffered with depth-2 prefetch: tile t+2 is issued while tile t
// computes, giving each async copy ~2 tiles of latency cover. Each wave issues
// exactly 5 async loads per tile and async loads complete in order, so
// s_wait_asynccnt 5 waits for the older tile only.
__global__ void __launch_bounds__(256) kgemm(const _Float16* __restrict__ Apack,
                                             const _Float16* __restrict__ Bpack,
                                             int* __restrict__ qn, float* __restrict__ cwm,
                                             float* __restrict__ topv) {
    __shared__ __attribute__((aligned(64))) _Float16 sB[3][TILE_HALFS]; // 3 x 20KB

    int bq   = blockIdx.x;
    int b    = bq / QQ;
    int wave = threadIdx.x >> 5;
    int lane = threadIdx.x & 31;
    int mtile = wave;

    const _Float16* Ap = Apack + ((size_t)bq * MT + mtile) * KST * FRAG + lane * 16;
    const _Float16* Bbase = Bpack + (size_t)b * NTILE * TILE_HALFS;

    const unsigned thOff = (unsigned)threadIdx.x * 16u;   // 256 thr x 5 x 16B = 20KB

    // issue async loads of column tile t into LDS buffer c (all 256 threads)
    auto prefetch_tile = [&](int t, int c) {
        unsigned long long gbase = (unsigned long long)(const void*)(Bbase + (size_t)t * TILE_HALFS);
        unsigned ldsaddr = (unsigned)(size_t)&sB[c][0] + thOff;
        unsigned goff    = thOff;
#pragma unroll
        for (int i = 0; i < 5; ++i) {
            asm volatile("global_load_async_to_lds_b128 %0, %1, %2"
                         :: "v"(ldsaddr + (unsigned)(i * 4096)),
                            "v"(goff + (unsigned)(i * 4096)),
                            "s"(gbase)
                         : "memory");
        }
    };

    // lane-local running stats (each lane owns a fixed column residue)
    float rmax[8]; int rarg[8];
    float t0[8], t1[8], t2[8];
#pragma unroll
    for (int r = 0; r < 8; ++r) { rmax[r] = NEG_BIG; rarg[r] = 0; }

    prefetch_tile(0, 0);
    prefetch_tile(1, 1);

    int t = 0;
    for (int n = 0; n < NWAY; ++n) {
#pragma unroll
        for (int r = 0; r < 8; ++r) { t0[r] = NEG_BIG; t1[r] = NEG_BIG; t2[r] = NEG_BIG; }

        for (int jt = 0; jt < JT; ++jt, ++t) {
            int cur = t % 3;
            // wait for tile t (older 5 loads); tile t+1's 5 may stay in flight
            if (t + 1 < NTILE) asm volatile("s_wait_asynccnt 0x5" ::: "memory");
            else               asm volatile("s_wait_asynccnt 0x0" ::: "memory");
            __syncthreads();   // all waves' portions landed; buffer (t+2)%3 free
            if (t + 2 < NTILE) prefetch_tile(t + 2, (t + 2) % 3);

            const _Float16* Bt = &sB[cur][lane * 16];
            // two accumulators to break the serial WMMA dependency chain
            v8f acc0 = {}, acc1 = {};
#pragma unroll
            for (int k = 0; k < KST; k += 2) {
                v16h a0 = *(const v16h*)(Ap + (size_t)k * FRAG);
                v16h b0 = *(const v16h*)(Bt + (size_t)k * FRAG);
                acc0 = __builtin_amdgcn_wmma_f32_16x16x32_f16(
                    false, a0, false, b0, (short)0, acc0, false, false);
                v16h a1 = *(const v16h*)(Ap + (size_t)(k + 1) * FRAG);
                v16h b1 = *(const v16h*)(Bt + (size_t)(k + 1) * FRAG);
                acc1 = __builtin_amdgcn_wmma_f32_16x16x32_f16(
                    false, a1, false, b1, (short)0, acc1, false, false);
            }
            v8f acc = acc0 + acc1;

            int  colc = jt * 16 + (lane & 15);
            bool vcol = colc < MS;
            int  jm   = n * MS + colc;      // merged (class-major) index for argmax
#pragma unroll
            for (int r = 0; r < 8; ++r) {
                float v = vcol ? acc[r] : NEG_BIG;
                if (v > rmax[r] || (v == rmax[r] && jm < rarg[r])) { rmax[r] = v; rarg[r] = jm; }
                top3_insert(v, t0[r], t1[r], t2[r]);
            }
        }
        // merge per-lane top-3 summaries across the 16 lanes of each half-wave
#pragma unroll
        for (int mk = 1; mk <= 8; mk <<= 1) {
#pragma unroll
            for (int r = 0; r < 8; ++r) {
                float o0 = __shfl_xor(t0[r], mk, 32);
                float o1 = __shfl_xor(t1[r], mk, 32);
                float o2 = __shfl_xor(t2[r], mk, 32);
                top3_insert(o0, t0[r], t1[r], t2[r]);
                top3_insert(o1, t0[r], t1[r], t2[r]);
                top3_insert(o2, t0[r], t1[r], t2[r]);
            }
        }
        if ((lane & 15) == 0) {
            int rbase = (lane >> 4) * 8;
#pragma unroll
            for (int r = 0; r < 8; ++r) {
                int mr = mtile * 16 + rbase + r;
                topv[((size_t)bq * 128 + mr) * NWAY + n] = (t0[r] + t1[r] + t2[r]) * (1.0f / 3.0f);
            }
        }
    }

    // single butterfly merge of the lane-local argmax (ties -> smaller j,
    // matching jnp.argmax first-occurrence semantics)
#pragma unroll
    for (int mk = 1; mk <= 8; mk <<= 1) {
#pragma unroll
        for (int r = 0; r < 8; ++r) {
            float ov = __shfl_xor(rmax[r], mk, 32);
            int   oj = __shfl_xor(rarg[r], mk, 32);
            if (ov > rmax[r] || (ov == rmax[r] && oj < rarg[r])) { rmax[r] = ov; rarg[r] = oj; }
        }
    }
    if ((lane & 15) == 0) {
        int rbase = (lane >> 4) * 8;
#pragma unroll
        for (int r = 0; r < 8; ++r) {
            int mr = mtile * 16 + rbase + r;
            qn[bq * 128 + mr]  = rarg[r];
            cwm[bq * 128 + mr] = rmax[r] + 1.0f;
        }
    }
}

// ---------------- mutual-NN mask + NBNN logits + per-episode CE ----------------
__global__ void kmask(const int* __restrict__ qn, const float* __restrict__ cwm,
                      const float* __restrict__ topv, const int* __restrict__ qy,
                      float* __restrict__ lossv) {
    __shared__ int   sqn[HW];
    __shared__ float scwm[HW];
    __shared__ float stv[HW * NWAY];
    __shared__ float contrib[128 * NWAY];
    __shared__ float qv[NWAY];
    int bq = blockIdx.x;
    int m  = threadIdx.x;   // 0..127
    if (m < HW) {
        sqn[m]  = qn[bq * 128 + m];
        scwm[m] = cwm[bq * 128 + m];
        for (int n = 0; n < NWAY; ++n) stv[m * NWAY + n] = topv[((size_t)bq * 128 + m) * NWAY + n];
    }
    __syncthreads();
    bool win = false;
    if (m < HW) {
        win = true;
        int myj = sqn[m]; float myc = scwm[m];
        for (int mp = 0; mp < HW; ++mp) {
            if (mp == m) continue;
            if (sqn[mp] == myj && (scwm[mp] > myc || (scwm[mp] == myc && mp < m))) { win = false; break; }
        }
    }
    for (int n = 0; n < NWAY; ++n)
        contrib[m * NWAY + n] = (win && m < HW) ? stv[m * NWAY + n] : 0.f;
    __syncthreads();
    if (m < NWAY) {
        float s = 0.f;
        for (int mp = 0; mp < HW; ++mp) s += contrib[mp * NWAY + m];
        qv[m] = s * 0.5f;                 // / TEMPERATURE (2.0)
    }
    __syncthreads();
    if (m == 0) {
        float mx = qv[0];
        for (int n = 1; n < NWAY; ++n) mx = fmaxf(mx, qv[n]);
        float se = 0.f;
        for (int n = 0; n < NWAY; ++n) se += expf(qv[n] - mx);
        float lse = logf(se) + mx;
        lossv[bq] = lse - qv[qy[bq]];
    }
}

__global__ void kfinal(const float* __restrict__ lossv, float* __restrict__ out) {
    __shared__ float red[128];
    int t = threadIdx.x;
    float s = 0.f;
    for (int i = t; i < BB * QQ; i += 128) s += lossv[i];
    red[t] = s;
    __syncthreads();
    for (int st = 64; st > 0; st >>= 1) {
        if (t < st) red[t] += red[t + st];
        __syncthreads();
    }
    if (t == 0) out[0] = red[0] / (float)(BB * QQ);
}

// ---------------- host launch ----------------
extern "C" void kernel_launch(void* const* d_in, const int* in_sizes, int n_in,
                              void* d_out, int out_size, void* d_ws, size_t ws_size,
                              hipStream_t stream) {
    (void)in_sizes; (void)n_in; (void)out_size; (void)ws_size;
    const float* support = (const float*)d_in[0];
    const float* query   = (const float*)d_in[1];
    const int*   qy      = (const int*)d_in[3];   // support_y (d_in[2]) unused

    char* ws = (char*)d_ws;
    size_t off = 0;
    float*    qscale = (float*)(ws + off);    off += (size_t)BB * QQ * 128 * 4;
    float*    sscale = (float*)(ws + off);    off += (size_t)BB * NWAY * MSP * 4;
    _Float16* Apack  = (_Float16*)(ws + off); off += (size_t)BB * QQ * MT * KST * FRAG * 2;
    _Float16* Bpack  = (_Float16*)(ws + off); off += (size_t)BB * NTILE * TILE_HALFS * 2;
    int*      qnbuf  = (int*)(ws + off);      off += (size_t)BB * QQ * 128 * 4;
    float*    cwmbuf = (float*)(ws + off);    off += (size_t)BB * QQ * 128 * 4;
    float*    topv   = (float*)(ws + off);    off += (size_t)BB * QQ * 128 * NWAY * 4;
    float*    lossv  = (float*)(ws + off);    off += (size_t)BB * QQ * 4;

    knorm_q<<<BB * QQ, 128, 0, stream>>>(query, qscale);
    knorm_s<<<BB * NWAY * KSHOT, 128, 0, stream>>>(support, sscale);
    kpackA<<<(BB * QQ * MT * KST * 32) / 256, 256, 0, stream>>>(query, qscale, Apack);
    kpackB<<<(BB * NWAY * JT * KST * 32) / 256, 256, 0, stream>>>(support, sscale, Bpack);
    kgemm<<<BB * QQ, 256, 0, stream>>>(Apack, Bpack, qnbuf, cwmbuf, topv);
    kmask<<<BB * QQ, 128, 0, stream>>>(qnbuf, cwmbuf, topv, qy, lossv);
    kfinal<<<1, 128, 0, stream>>>(lossv, (float*)d_out);
}
